// ADMETNet_27822798143741
// MI455X (gfx1250) — compile-verified, matrix operand order checked by reading, not run on hardware
//
#include <hip/hip_runtime.h>

#define N_NODES   200000
#define N_EDGES   860000
#define BGRAPHS   4096
#define NDIM      128
#define EDIM      64
#define NHEADS    4
#define NLAYERS   4
#define FPDIM     2048
#define FPEMB     256
#define SHDIM     512
#define NTASKS    12
#define ATOMF     74
#define BONDF     12
#define LN_EPS    1e-5f
#define NEG_INF_KEY 0x007FFFFFu   // fkey(-inf)

typedef __attribute__((ext_vector_type(16))) _Float16 v16h;
typedef __attribute__((ext_vector_type(8)))  float    v8f;
typedef __attribute__((ext_vector_type(4)))  _Float16 h4_t;

// ---------------------------------------------------------------- helpers
__device__ __forceinline__ float gelu_f(float v) {
    return 0.5f * v * (1.0f + erff(v * 0.70710678118654752440f));
}
// monotonic float<->uint key for atomic max on floats (handles negatives)
__device__ __forceinline__ unsigned fkey(float f) {
    unsigned u = __float_as_uint(f);
    return (u & 0x80000000u) ? ~u : (u | 0x80000000u);
}
__device__ __forceinline__ float fdecode(unsigned k) {
    unsigned u = (k & 0x80000000u) ? (k & 0x7FFFFFFFu) : ~k;
    return __uint_as_float(u);
}

// ---------------------------------------------------------------- fill
__global__ void admet_fill(unsigned* __restrict__ p, unsigned v, long n) {
    long i = (long)blockIdx.x * blockDim.x + threadIdx.x;
    if (i < n) p[i] = v;
}

// ---------------------------------------------------------------- WMMA GEMM
// C[M,N](ldc) = A[M,K] @ B[K,N] (+bias[N]); fp32 in/out, f16 WMMA, f32 accum.
// blockDim=128 (4 waves). Block tile 64x64; each wave: 32x32 via 2x2 WMMA.
// gridDim.z batches with per-batch pointer strides (sA may be 0 = shared A).
// REQUIRES N % 64 == 0 (true for every GEMM in this net).
// B is staged TRANSPOSED into LDS so both A and B fragments are two
// contiguous 16B runs per lane -> ds_load_b128 feeds the WMMAs.
// KVEC template (K%4==0) removes the scalar staging fallback from the hot
// instantiation; only the K=74 atom-embed GEMM uses KVEC=false.
template <bool KVEC>
__global__ __launch_bounds__(128) void admet_gemm(
    const float* __restrict__ A, const float* __restrict__ B,
    const float* __restrict__ bias, float* __restrict__ C,
    int M, int N, int K, int ldc,
    long sA, long sB, long sBias, long sC)
{
    __shared__ _Float16 As [64][40];  // [M][K]  80B row stride
    __shared__ _Float16 Bst[64][40];  // [N][K]  (transposed) 80B row stride

    long bz = blockIdx.z;
    A += bz * sA;
    B += bz * sB;
    C += bz * sC;
    if (bias) bias += bz * sBias;

    const int m0   = blockIdx.x * 64;
    const int n0   = blockIdx.y * 64;
    const int tid  = threadIdx.x;
    const int lane = tid & 31;
    const int wave = tid >> 5;
    const int wm   = (wave >> 1) * 32;   // wave M offset in tile
    const int wn   = (wave & 1)  * 32;   // wave N offset in tile
    const int fm   = lane & 15;
    const int hi   = lane >> 4;

    v8f acc[2][2] = {};

    for (int k0 = 0; k0 < K; k0 += 32) {
        const bool fullK = (k0 + 32 <= K);   // uniform per tile

        if (KVEC) {
            // ---- stage A tile (64 M x 32 K): float4 -> half4 ----
            #pragma unroll
            for (int j = 0; j < 4; ++j) {
                int idx = tid + j * 128;            // 512 chunks
                int r = idx >> 3, c4 = idx & 7;     // 8 chunks per M row
                int gm = m0 + r, gk = k0 + c4 * 4;
                float4 v = make_float4(0.f, 0.f, 0.f, 0.f);
                bool ok = fullK ? (gm < M) : (gm < M && gk + 3 < K);
                if (ok) v = *(const float4*)(A + (long)gm * K + gk);
                h4_t h;
                h[0] = (_Float16)v.x; h[1] = (_Float16)v.y;
                h[2] = (_Float16)v.z; h[3] = (_Float16)v.w;
                *(h4_t*)&As[r][c4 * 4] = h;
            }
            // ---- stage B tile transposed (64 N x 32 K) ----
            // per chunk: 4 K-consecutive scalars (each coalesced across lanes
            // along N) -> one ds_store_b64
            #pragma unroll
            for (int j = 0; j < 4; ++j) {
                int idx = tid + j * 128;            // 512 chunks
                int n = idx & 63, kc = idx >> 6;    // 8 chunks per N row
                int gn = n0 + n;                    // always < N (N%64==0)
                float vs[4] = {0.f, 0.f, 0.f, 0.f};
                if (fullK) {
                    #pragma unroll
                    for (int t = 0; t < 4; ++t)
                        vs[t] = B[(long)(k0 + kc * 4 + t) * N + gn];
                } else {
                    #pragma unroll
                    for (int t = 0; t < 4; ++t) {
                        int gk = k0 + kc * 4 + t;
                        if (gk < K) vs[t] = B[(long)gk * N + gn];
                    }
                }
                h4_t h;
                h[0] = (_Float16)vs[0]; h[1] = (_Float16)vs[1];
                h[2] = (_Float16)vs[2]; h[3] = (_Float16)vs[3];
                *(h4_t*)&Bst[n][kc * 4] = h;
            }
        } else {
            // scalar fallback (K % 4 != 0; only the K=74 GEMM)
            for (int i = tid; i < 64 * 32; i += 128) {
                int r = i >> 5, c = i & 31;
                int gm = m0 + r, gk = k0 + c;
                float v = (gm < M && gk < K) ? A[(long)gm * K + gk] : 0.0f;
                As[r][c] = (_Float16)v;
            }
            for (int i = tid; i < 32 * 64; i += 128) {
                int kk = i >> 6, n = i & 63;
                int gk = k0 + kk, gn = n0 + n;
                float v = (gk < K) ? B[(long)gk * N + gn] : 0.0f;
                Bst[n][kk] = (_Float16)v;
            }
        }
        __syncthreads();

        // fragments: per lane, elements 0..7 = K[hi*8 .. hi*8+7],
        // elements 8..15 = K[16+hi*8 .. 16+hi*8+7] -> 2x b128 each
        v16h afrag[2], bfrag[2];
        #pragma unroll
        for (int half = 0; half < 2; ++half) {
            int arow = wm + half * 16 + fm;
            int brow = wn + half * 16 + fm;
            #pragma unroll
            for (int e = 0; e < 8; ++e) {
                afrag[half][e]     = As [arow][hi * 8 + e];
                afrag[half][e + 8] = As [arow][16 + hi * 8 + e];
                bfrag[half][e]     = Bst[brow][hi * 8 + e];
                bfrag[half][e + 8] = Bst[brow][16 + hi * 8 + e];
            }
        }
        #pragma unroll
        for (int mi = 0; mi < 2; ++mi)
            #pragma unroll
            for (int ni = 0; ni < 2; ++ni)
                acc[mi][ni] = __builtin_amdgcn_wmma_f32_16x16x32_f16(
                    false, afrag[mi], false, bfrag[ni],
                    (short)0, acc[mi][ni], false, false);
        __syncthreads();
    }

    // store: C/D layout — VGPR r: lanes0-15 M=r, lanes16-31 M=r+8; N=lane&15
    #pragma unroll
    for (int mi = 0; mi < 2; ++mi)
        #pragma unroll
        for (int ni = 0; ni < 2; ++ni)
            #pragma unroll
            for (int r = 0; r < 8; ++r) {
                int gm = m0 + wm + mi * 16 + r + hi * 8;
                int gn = n0 + wn + ni * 16 + fm;
                if (gm < M) {
                    float v = acc[mi][ni][r];
                    if (bias) v += bias[gn];
                    C[(long)gm * ldc + gn] = v;
                }
            }
}

// ---------------------------------------------------------------- LayerNorm+GELU
// one block per row; task_rows>0 -> per-task gamma/beta (g += (row/task_rows)*N)
__global__ __launch_bounds__(128) void admet_ln_gelu(
    float* __restrict__ X, const float* __restrict__ g, const float* __restrict__ b,
    int M, int N, int task_rows)
{
    int row = blockIdx.x;
    if (row >= M) return;
    const float* gp = g;
    const float* bp = b;
    if (task_rows > 0) {
        int t = row / task_rows;
        gp += (long)t * N;
        bp += (long)t * N;
    }
    float* xr = X + (long)row * N;
    float s = 0.f, s2 = 0.f;
    for (int c = threadIdx.x; c < N; c += 128) {
        float v = xr[c];
        s += v; s2 += v * v;
    }
    #pragma unroll
    for (int off = 1; off < 32; off <<= 1) {
        s  += __shfl_xor(s, off, 32);
        s2 += __shfl_xor(s2, off, 32);
    }
    __shared__ float ls[4], ls2[4];
    int wave = threadIdx.x >> 5, lane = threadIdx.x & 31;
    if (lane == 0) { ls[wave] = s; ls2[wave] = s2; }
    __syncthreads();
    s  = ls[0] + ls[1] + ls[2] + ls[3];
    s2 = ls2[0] + ls2[1] + ls2[2] + ls2[3];
    float mean = s / N;
    float var  = s2 / N - mean * mean;
    float inv  = rsqrtf(var + LN_EPS);
    for (int c = threadIdx.x; c < N; c += 128) {
        float v = (xr[c] - mean) * inv * gp[c] + bp[c];
        xr[c] = gelu_f(v);
    }
}

// ---------------------------------------------------------------- BatchNorm (128 cols)
__global__ __launch_bounds__(128) void admet_bn_stats(
    const float* __restrict__ X, float* __restrict__ sum, float* __restrict__ sumsq, int M)
{
    int c = threadIdx.x;
    long r0 = (long)blockIdx.x * 256;
    long r1 = r0 + 256; if (r1 > M) r1 = M;
    float s = 0.f, s2 = 0.f;
    for (long r = r0; r < r1; ++r) {
        float v = X[r * 128 + c];
        s += v; s2 += v * v;
    }
    atomicAdd(&sum[c], s);
    atomicAdd(&sumsq[c], s2);
}

__global__ void admet_bn_res_gelu(
    const float* __restrict__ agg, float* __restrict__ h,
    const float* __restrict__ sum, const float* __restrict__ sumsq,
    const float* __restrict__ g, const float* __restrict__ b, long total)
{
    long i = (long)blockIdx.x * blockDim.x + threadIdx.x;
    if (i >= total) return;
    int c = (int)(i & 127);
    float mean = sum[c] * (1.0f / N_NODES);
    float var  = sumsq[c] * (1.0f / N_NODES) - mean * mean;
    float v = (agg[i] - mean) * rsqrtf(var + LN_EPS) * g[c] + b[c];
    h[i] = gelu_f(v + h[i]);   // residual + GELU
}

// ---------------------------------------------------------------- GAT attention
__global__ __launch_bounds__(256) void admet_gat_score(
    const float* __restrict__ xl, const float* __restrict__ xr, const float* __restrict__ xe,
    const int* __restrict__ src, const int* __restrict__ dst, const float* __restrict__ att,
    float* __restrict__ s, unsigned* __restrict__ smaxk, long totalEH)
{
    long i = (long)blockIdx.x * 256 + threadIdx.x;
    if (i >= totalEH) return;
    long e = i >> 2;
    int  h = (int)(i & 3);
    int sn = src[e], dn = dst[e];
    const float4* a4 = (const float4*)(xl + (long)sn * 128 + h * 32);
    const float4* r4 = (const float4*)(xr + (long)dn * 128 + h * 32);
    const float4* q4 = (const float4*)(xe + e * 128 + h * 32);
    const float4* t4 = (const float4*)(att + h * 32);
    float acc = 0.f;
    #pragma unroll
    for (int c = 0; c < 8; ++c) {
        float4 va = a4[c], vr = r4[c], vq = q4[c], vt = t4[c];
        float m;
        m = va.x + vr.x + vq.x; m = (m > 0.f) ? m : 0.2f * m; acc += m * vt.x;
        m = va.y + vr.y + vq.y; m = (m > 0.f) ? m : 0.2f * m; acc += m * vt.y;
        m = va.z + vr.z + vq.z; m = (m > 0.f) ? m : 0.2f * m; acc += m * vt.z;
        m = va.w + vr.w + vq.w; m = (m > 0.f) ? m : 0.2f * m; acc += m * vt.w;
    }
    s[i] = acc;
    atomicMax(&smaxk[(long)dn * 4 + h], fkey(acc));
}

__global__ void admet_smax_fin(const unsigned* __restrict__ k, float* __restrict__ smax, long n)
{
    long i = (long)blockIdx.x * blockDim.x + threadIdx.x;
    if (i >= n) return;
    float v = fdecode(k[i]);
    if (!isfinite(v)) v = 0.f;   // empty segment -> 0 (matches reference where())
    smax[i] = v;
}

__global__ void admet_gat_ex(
    const float* __restrict__ s, const float* __restrict__ smax,
    const int* __restrict__ dst, float* __restrict__ ex, float* __restrict__ den, long totalEH)
{
    long i = (long)blockIdx.x * blockDim.x + threadIdx.x;
    if (i >= totalEH) return;
    long e = i >> 2;
    int  h = (int)(i & 3);
    int dn = dst[e];
    float v = expf(s[i] - smax[(long)dn * 4 + h]);
    ex[i] = v;
    atomicAdd(&den[(long)dn * 4 + h], v);
}

__global__ void admet_gat_agg(
    const float* __restrict__ ex, const float* __restrict__ den, const float* __restrict__ xl,
    const int* __restrict__ src, const int* __restrict__ dst, float* __restrict__ agg, long totalEC)
{
    long i = (long)blockIdx.x * blockDim.x + threadIdx.x;
    if (i >= totalEC) return;            // E * 32 groups (4 channels each)
    long e = i >> 5;
    int  q = (int)(i & 31);
    int  h = q >> 3;
    int sn = src[e], dn = dst[e];
    float alpha = ex[e * 4 + h] / (den[(long)dn * 4 + h] + 1e-16f);
    float4 v = *(const float4*)(xl + (long)sn * 128 + q * 4);
    float* o = agg + (long)dn * 128 + q * 4;
    atomicAdd(o + 0, alpha * v.x);
    atomicAdd(o + 1, alpha * v.y);
    atomicAdd(o + 2, alpha * v.z);
    atomicAdd(o + 3, alpha * v.w);
}

// ---------------------------------------------------------------- pooling
__global__ void admet_count(const int* __restrict__ batch, float* __restrict__ cnt, int n)
{
    int i = blockIdx.x * blockDim.x + threadIdx.x;
    if (i < n) atomicAdd(&cnt[batch[i]], 1.0f);
}

__global__ void admet_pool_accum(
    const float* __restrict__ h, const int* __restrict__ batch,
    float* __restrict__ psum, unsigned* __restrict__ pmax, long total)
{
    long i = (long)blockIdx.x * blockDim.x + threadIdx.x;
    if (i >= total) return;
    long n = i >> 7;
    int  c = (int)(i & 127);
    int  b = batch[n];
    float v = h[i];
    atomicAdd(&psum[(long)b * 128 + c], v);
    atomicMax(&pmax[(long)b * 128 + c], fkey(v));
}

__global__ void admet_pool_fin(
    const float* __restrict__ psum, const unsigned* __restrict__ pmax,
    const float* __restrict__ cnt, float* __restrict__ fin, long total)
{
    long i = (long)blockIdx.x * blockDim.x + threadIdx.x;
    if (i >= total) return;              // BGRAPHS * 128
    long b = i >> 7;
    int  c = (int)(i & 127);
    float m  = psum[i] / fmaxf(cnt[b], 1.0f);
    float mx = fdecode(pmax[i]);
    if (!isfinite(mx)) mx = 0.f;
    fin[b * 512 + c]       = m;          // mean pool -> cols [0,128)
    fin[b * 512 + 128 + c] = mx;         // max pool  -> cols [128,256)
}

// ---------------------------------------------------------------- final head dot
__global__ void admet_head_out(
    const float* __restrict__ t2, const float* __restrict__ Wo,
    const float* __restrict__ bo, float* __restrict__ out, int total)
{
    int i = blockIdx.x * blockDim.x + threadIdx.x;
    if (i >= total) return;              // NTASKS * BGRAPHS
    int t = i >> 12;                     // / 4096
    const float4* x4 = (const float4*)(t2 + (long)i * 128);
    const float4* w4 = (const float4*)(Wo + (long)t * 128);
    float acc = 0.f;
    #pragma unroll
    for (int c = 0; c < 32; ++c) {
        float4 xv = x4[c], wv = w4[c];
        acc += xv.x * wv.x + xv.y * wv.y + xv.z * wv.z + xv.w * wv.w;
    }
    out[i] = acc + bo[t];
}

// ================================================================ host
struct IdxMap {
    int x, edge_index, edge_attr, fp, batch;
    int atomW, atomB, atomG, atomBt;
    int bondW, bondB, bondG, bondBt;
    int lWl[4], lBl[4], lWr[4], lBr[4], lWe[4], lAtt[4], lBnG[4], lBnB[4];
    int fpW, fpB, fpG, fpBt, fpWo, fpBo;
    int fuW, fuB, fuG, fuBt, fuWo, fuBo;
    int hW1, hB1, hG1, hBt1, hW2, hB2, hG2, hBt2, hWo, hBo;
};

extern "C" void kernel_launch(void* const* d_in, const int* in_sizes, int n_in,
                              void* d_out, int out_size, void* d_ws, size_t ws_size,
                              hipStream_t stream)
{
    IdxMap m;
    if (in_sizes[0] == N_NODES * ATOMF) {
        // Python dict insertion-order flattening
        m.x = 0; m.edge_index = 1; m.edge_attr = 2; m.fp = 3; m.batch = 4;
        m.atomW = 5; m.atomB = 6; m.atomG = 7; m.atomBt = 8;
        m.bondW = 9; m.bondB = 10; m.bondG = 11; m.bondBt = 12;
        for (int l = 0; l < NLAYERS; ++l) {
            int base = 13 + 9 * l;
            m.lWl[l] = base + 0; m.lBl[l] = base + 1; m.lWr[l] = base + 2;
            m.lBr[l] = base + 3; m.lWe[l] = base + 4; m.lAtt[l] = base + 5;
            /* bias @ +6 cancelled by BN */ m.lBnG[l] = base + 7; m.lBnB[l] = base + 8;
        }
        m.fpW = 49; m.fpB = 50; m.fpG = 51; m.fpBt = 52; m.fpWo = 53; m.fpBo = 54;
        m.fuW = 55; m.fuB = 56; m.fuG = 57; m.fuBt = 58; m.fuWo = 59; m.fuBo = 60;
        m.hW1 = 61; m.hB1 = 62; m.hG1 = 63; m.hBt1 = 64;
        m.hW2 = 65; m.hB2 = 66; m.hG2 = 67; m.hBt2 = 68; m.hWo = 69; m.hBo = 70;
    } else {
        // jax tree-flatten (dict keys sorted) flattening
        m.batch = 0; m.edge_attr = 1; m.edge_index = 2; m.fp = 3;
        m.atomW = 4; m.atomB = 5; m.atomBt = 6; m.atomG = 7;
        m.bondW = 8; m.bondB = 9; m.bondBt = 10; m.bondG = 11;
        m.fpWo = 12; m.fpBo = 13; m.fpW = 14; m.fpB = 15; m.fpBt = 16; m.fpG = 17;
        m.fuWo = 18; m.fuBo = 19; m.fuW = 20; m.fuB = 21; m.fuBt = 22; m.fuG = 23;
        m.hW1 = 24; m.hW2 = 25; m.hWo = 26; m.hB1 = 27; m.hB2 = 28; m.hBo = 29;
        m.hBt1 = 30; m.hBt2 = 31; m.hG1 = 32; m.hG2 = 33;
        for (int l = 0; l < NLAYERS; ++l) {
            int base = 34 + 9 * l;  // keys: We,Wl,Wr,att,bias,bl,bn_b,bn_g,br
            m.lWe[l] = base + 0; m.lWl[l] = base + 1; m.lWr[l] = base + 2;
            m.lAtt[l] = base + 3; /* bias +4 */ m.lBl[l] = base + 5;
            m.lBnB[l] = base + 6; m.lBnG[l] = base + 7; m.lBr[l] = base + 8;
        }
        m.x = 70;
    }

    auto F = [&](int i) { return (const float*)d_in[i]; };
    auto I = [&](int i) { return (const int*)d_in[i]; };

    const float* x_in  = F(m.x);
    const int*   src   = I(m.edge_index);
    const int*   dst   = I(m.edge_index) + N_EDGES;
    const float* eattr = F(m.edge_attr);
    const float* fp_in = F(m.fp);
    const int*   batch = I(m.batch);

    // ---------------- workspace bump allocator (256B aligned) ----------------
    char* wp = (char*)d_ws;
    auto alloc = [&](size_t bytes) -> void* {
        void* p = (void*)wp;
        wp += (bytes + 255) & ~(size_t)255;
        return p;
    };
    float*    f_h     = (float*)alloc((size_t)N_NODES * 128 * 4);
    float*    f_xl    = (float*)alloc((size_t)N_NODES * 128 * 4);
    float*    f_xr    = (float*)alloc((size_t)N_NODES * 128 * 4);
    float*    f_agg   = (float*)alloc((size_t)N_NODES * 128 * 4);
    float*    f_ea    = (float*)alloc((size_t)N_EDGES * 64 * 4);
    float*    f_xe    = (float*)alloc((size_t)N_EDGES * 128 * 4);
    float*    f_s     = (float*)alloc((size_t)N_EDGES * 4 * 4);
    float*    f_ex    = (float*)alloc((size_t)N_EDGES * 4 * 4);
    unsigned* u_smaxk = (unsigned*)alloc((size_t)N_NODES * 4 * 4);
    float*    f_smax  = (float*)alloc((size_t)N_NODES * 4 * 4);
    float*    f_den   = (float*)alloc((size_t)N_NODES * 4 * 4);
    float*    f_bnsum = (float*)alloc(128 * 4);
    float*    f_bnsq  = (float*)alloc(128 * 4);
    float*    f_cnt   = (float*)alloc((size_t)BGRAPHS * 4);
    float*    f_psum  = (float*)alloc((size_t)BGRAPHS * 128 * 4);
    unsigned* u_pmaxk = (unsigned*)alloc((size_t)BGRAPHS * 128 * 4);
    float*    f_fin   = (float*)alloc((size_t)BGRAPHS * 512 * 4);  // [mean|max|fp_emb]
    float*    f_fphid = (float*)alloc((size_t)BGRAPHS * 512 * 4);
    float*    f_fush  = (float*)alloc((size_t)BGRAPHS * 512 * 4);
    float*    f_fused = (float*)alloc((size_t)BGRAPHS * 512 * 4);
    float*    f_t1    = (float*)alloc((size_t)NTASKS * BGRAPHS * 256 * 4);
    float*    f_t2    = (float*)alloc((size_t)NTASKS * BGRAPHS * 128 * 4);
    if ((size_t)(wp - (char*)d_ws) > ws_size) return;  // scratch too small

    // ---------------- launch helpers ----------------
    auto gemm = [&](const float* A, const float* B, const float* bias, float* C,
                    int M, int N, int K, int ldc, int nb,
                    long sA, long sB, long sBias, long sC) {
        dim3 g((M + 63) / 64, (N + 63) / 64, nb);
        if ((K & 3) == 0)
            admet_gemm<true><<<g, 128, 0, stream>>>(A, B, bias, C, M, N, K, ldc, sA, sB, sBias, sC);
        else
            admet_gemm<false><<<g, 128, 0, stream>>>(A, B, bias, C, M, N, K, ldc, sA, sB, sBias, sC);
    };
    auto lngelu = [&](float* X, const float* g, const float* b, int M, int N, int tr) {
        admet_ln_gelu<<<M, 128, 0, stream>>>(X, g, b, M, N, tr);
    };
    auto fill = [&](void* p, unsigned v, long n) {
        admet_fill<<<(unsigned)((n + 255) / 256), 256, 0, stream>>>((unsigned*)p, v, n);
    };

    // ================ 1) node & edge embeddings ================
    gemm(x_in, F(m.atomW), F(m.atomB), f_h, N_NODES, NDIM, ATOMF, NDIM, 1, 0, 0, 0, 0);
    lngelu(f_h, F(m.atomG), F(m.atomBt), N_NODES, NDIM, 0);
    gemm(eattr, F(m.bondW), F(m.bondB), f_ea, N_EDGES, EDIM, BONDF, EDIM, 1, 0, 0, 0, 0);
    lngelu(f_ea, F(m.bondG), F(m.bondBt), N_EDGES, EDIM, 0);

    // ================ 2) GATv2 layers ================
    const long EH = (long)N_EDGES * 4;
    const long EC = (long)N_EDGES * 32;
    const long NC = (long)N_NODES * 128;
    for (int l = 0; l < NLAYERS; ++l) {
        gemm(f_h, F(m.lWl[l]), F(m.lBl[l]), f_xl, N_NODES, NDIM, NDIM, NDIM, 1, 0, 0, 0, 0);
        gemm(f_h, F(m.lWr[l]), F(m.lBr[l]), f_xr, N_NODES, NDIM, NDIM, NDIM, 1, 0, 0, 0, 0);
        gemm(f_ea, F(m.lWe[l]), nullptr,    f_xe, N_EDGES, NDIM, EDIM, NDIM, 1, 0, 0, 0, 0);

        fill(u_smaxk, NEG_INF_KEY, (long)N_NODES * 4);
        fill(f_den,   0u,          (long)N_NODES * 4);
        fill(f_agg,   0u,          NC);
        fill(f_bnsum, 0u, 128);
        fill(f_bnsq,  0u, 128);

        admet_gat_score<<<(unsigned)((EH + 255) / 256), 256, 0, stream>>>(
            f_xl, f_xr, f_xe, src, dst, F(m.lAtt[l]), f_s, u_smaxk, EH);
        admet_smax_fin<<<(unsigned)((N_NODES * 4 + 255) / 256), 256, 0, stream>>>(
            u_smaxk, f_smax, (long)N_NODES * 4);
        admet_gat_ex<<<(unsigned)((EH + 255) / 256), 256, 0, stream>>>(
            f_s, f_smax, dst, f_ex, f_den, EH);
        admet_gat_agg<<<(unsigned)((EC + 255) / 256), 256, 0, stream>>>(
            f_ex, f_den, f_xl, src, dst, f_agg, EC);

        admet_bn_stats<<<(N_NODES + 255) / 256, 128, 0, stream>>>(f_agg, f_bnsum, f_bnsq, N_NODES);
        admet_bn_res_gelu<<<(unsigned)((NC + 255) / 256), 256, 0, stream>>>(
            f_agg, f_h, f_bnsum, f_bnsq, F(m.lBnG[l]), F(m.lBnB[l]), NC);
    }

    // ================ 3) graph pooling -> f_fin[:, 0:256] ================
    fill(f_cnt,   0u, BGRAPHS);
    fill(f_psum,  0u, (long)BGRAPHS * 128);
    fill(u_pmaxk, NEG_INF_KEY, (long)BGRAPHS * 128);
    admet_count<<<(N_NODES + 255) / 256, 256, 0, stream>>>(batch, f_cnt, N_NODES);
    admet_pool_accum<<<(unsigned)((NC + 255) / 256), 256, 0, stream>>>(f_h, batch, f_psum, u_pmaxk, NC);
    admet_pool_fin<<<(BGRAPHS * 128 + 255) / 256, 256, 0, stream>>>(
        f_psum, u_pmaxk, f_cnt, f_fin, (long)BGRAPHS * 128);

    // ================ 4) fingerprint MLP -> f_fin[:, 256:512] ================
    gemm(fp_in, F(m.fpW), F(m.fpB), f_fphid, BGRAPHS, 512, FPDIM, 512, 1, 0, 0, 0, 0);
    lngelu(f_fphid, F(m.fpG), F(m.fpBt), BGRAPHS, 512, 0);
    gemm(f_fphid, F(m.fpWo), F(m.fpBo), f_fin + 256, BGRAPHS, FPEMB, 512, 512, 1, 0, 0, 0, 0);

    // ================ 5) fusion MLP ================
    gemm(f_fin, F(m.fuW), F(m.fuB), f_fush, BGRAPHS, SHDIM, 512, SHDIM, 1, 0, 0, 0, 0);
    lngelu(f_fush, F(m.fuG), F(m.fuBt), BGRAPHS, 512, 0);
    gemm(f_fush, F(m.fuWo), F(m.fuBo), f_fused, BGRAPHS, SHDIM, SHDIM, SHDIM, 1, 0, 0, 0, 0);

    // ================ 6) 12 task heads (batched over gridDim.z) ================
    gemm(f_fused, F(m.hW1), F(m.hB1), f_t1, BGRAPHS, 256, SHDIM, 256,
         NTASKS, 0, 512L * 256, 256, (long)BGRAPHS * 256);
    lngelu(f_t1, F(m.hG1), F(m.hBt1), NTASKS * BGRAPHS, 256, BGRAPHS);
    gemm(f_t1, F(m.hW2), F(m.hB2), f_t2, BGRAPHS, 128, 256, 128,
         NTASKS, (long)BGRAPHS * 256, 256L * 128, 128, (long)BGRAPHS * 128);
    lngelu(f_t2, F(m.hG2), F(m.hBt2), NTASKS * BGRAPHS, 128, BGRAPHS);
    admet_head_out<<<(NTASKS * BGRAPHS + 255) / 256, 256, 0, stream>>>(
        f_t2, F(m.hWo), F(m.hBo), (float*)d_out, NTASKS * BGRAPHS);
}
